// AGNO_82575041233033
// MI455X (gfx1250) — matrix operand order repaired and checked by reading, not compiled, not run on hardware
//
#include <hip/hip_runtime.h>
#include <hip/hip_fp16.h>
#include <math.h>

typedef __attribute__((ext_vector_type(16))) _Float16 v16h;
typedef __attribute__((ext_vector_type(8)))  float    v8f;

union F16x16 { v16h v; unsigned u[8]; _Float16 h[16]; };

#define WAVES_PER_BLOCK 8
#define THREADS (WAVES_PER_BLOCK * 32)

// ---------------- LDS layout (bytes) ----------------
// Weight tiles are stored fragment-major: [tile][lane][16 halves] so each
// lane's WMMA B operand is one contiguous 32B read.
#define WB1_OFF   0                   // W1: 8 tiles  * 512 halves = 8192 B
#define WB2_OFF   8192                // W2: 32 tiles * 512 halves = 32768 B
#define WB3_OFF   40960               // W3: 16 tiles * 512 halves = 16384 B
#define B1_OFF    57344               // 128 f32
#define B2_OFF    57856               // 128 f32
#define B3_OFF    58368               // 64  f32
#define WSCR_OFF  58624               // per-wave scratch
#define WSCR_SIZE 4608                // inbuf 384 + attn 64 + nbr 64 + pad + stage 4096
#define SMEM_BYTES (WSCR_OFF + WAVES_PER_BLOCK * WSCR_SIZE)   // 95488

// A-matrix (16x32 f16) element map per CDNA5 ISA 7.12.2:
// lane holds row M = lane&15; half j of its v16h holds
// K = (vgprIdx>=4 ? 16:0) + (lane>=16 ? 8:0) + 2*(vgprIdx&3) + (j&1)
__device__ __forceinline__ int a_frag_k(int lane, int j) {
  int v = j >> 1, h = j & 1;
  return ((v & 4) << 2) + ((lane & 16) >> 1) + ((v & 3) << 1) + h;
}
// B-matrix (32x16 f16): same K map as A: half j holds
// K = (j>=8 ? 16:0) + (lane>=16 ? 8:0) + (j&7), column N = lane&15.
__device__ __forceinline__ int b_frag_k(int lane, int j) {
  return ((j & 8) << 1) + ((lane & 16) >> 1) + (j & 7);
}

__device__ __forceinline__ float gelu_exact(float x) {
  return 0.5f * x * (1.0f + erff(x * 0.70710678118654752440f));
}

extern "C" __global__ __launch_bounds__(THREADS)
void agno_fused_wmma(const float* __restrict__ y,
                     const float* __restrict__ f_y,
                     const int*   __restrict__ nidx,
                     const float* __restrict__ W1, const float* __restrict__ b1,
                     const float* __restrict__ W2, const float* __restrict__ b2,
                     const float* __restrict__ W3, const float* __restrict__ b3,
                     float* __restrict__ out, int n)
{
  extern __shared__ char smem[];
  const int tid  = threadIdx.x;
  const int lane = tid & 31;
  const int wave = tid >> 5;

  // ---- Stage weights into LDS in WMMA B-fragment layout (once per block) ----
  {
    __half* wh = (__half*)smem;   // all three weight regions are contiguous
    for (int idx = tid; idx < 28672; idx += THREADS) {
      int tile = idx >> 9;
      int e    = idx & 511;
      int l    = e >> 4;          // fragment lane
      int j    = e & 15;          // fragment half index
      int K    = b_frag_k(l, j);
      int Nc   = l & 15;
      float f;
      if (tile < 8) {             // W1: [6][128], K padded 6 -> 32 with zeros
        int Ks = (K < 6) ? K : 0;
        float w = W1[Ks * 128 + tile * 16 + Nc];
        f = (K < 6) ? w : 0.0f;
      } else if (tile < 40) {     // W2: [128][128], tiles [kt(4)][nt(8)]
        int t = tile - 8; int kt = t >> 3; int nt = t & 7;
        f = W2[(kt * 32 + K) * 128 + nt * 16 + Nc];
      } else {                    // W3: [128][64], tiles [kt(4)][nt(4)]
        int t = tile - 40; int kt = t >> 2; int nt = t & 3;
        f = W3[(kt * 32 + K) * 64 + nt * 16 + Nc];
      }
      wh[idx] = __float2half(f);
    }
    float* bb = (float*)(smem + B1_OFF);
    for (int i = tid; i < 320; i += THREADS)
      bb[i] = (i < 128) ? b1[i] : (i < 256) ? b2[i - 128] : b3[i - 256];
  }
  __syncthreads();

  const float* bias1 = (const float*)(smem + B1_OFF);
  const float* bias2 = (const float*)(smem + B2_OFF);
  const float* bias3 = (const float*)(smem + B3_OFF);

  char*   wscr  = smem + WSCR_OFF + wave * WSCR_SIZE;
  float*  inbuf = (float*)wscr;              // [16][6] edge inputs (rep||slf)
  float*  attnb = (float*)(wscr + 384);      // [16]
  int*    nbrb  = (int*)  (wscr + 448);      // [16]
  __half* hbuf  = (__half*)(wscr + 512);     // [16][128] f16 activations
  float*  kbuf  = (float*) (wscr + 512);     // [16][64]  f32 (aliases hbuf)

  const int M    = lane & 15;
  const int mrow = (lane & 16) ? 8 : 0;      // C-layout row base for this lane

  for (int node = blockIdx.x * WAVES_PER_BLOCK + wave; node < n;
       node += gridDim.x * WAVES_PER_BLOCK) {     // wave-uniform loop

    const float sx = y[node * 3 + 0];
    const float sy = y[node * 3 + 1];
    const float sz = y[node * 3 + 2];

    // ------- per-edge cosine score + segment softmax (lanes 0..15) -------
    if (lane < 16) {
      const int nb = nidx[node * 16 + lane];
      const float rx = y[nb * 3 + 0], ry = y[nb * 3 + 1], rz = y[nb * 3 + 2];
      float snorm = fmaxf(sqrtf(sx * sx + sy * sy + sz * sz), 1e-12f);
      float rnorm = fmaxf(sqrtf(rx * rx + ry * ry + rz * rz), 1e-12f);
      float score = (sx * rx + sy * ry + sz * rz) / (snorm * rnorm);
      float mx = score;
      #pragma unroll
      for (int off = 1; off < 16; off <<= 1)
        mx = fmaxf(mx, __shfl_xor(mx, off, 16));
      float ev = expf(score - mx);
      float ss = ev;
      #pragma unroll
      for (int off = 1; off < 16; off <<= 1)
        ss += __shfl_xor(ss, off, 16);
      attnb[lane] = ev / ss;
      nbrb[lane]  = nb;
      inbuf[lane * 6 + 0] = rx; inbuf[lane * 6 + 1] = ry; inbuf[lane * 6 + 2] = rz;
      inbuf[lane * 6 + 3] = sx; inbuf[lane * 6 + 4] = sy; inbuf[lane * 6 + 5] = sz;
    }

    // ------- Layer 1: [16x6 pad 32] @ [32x128] + b1, GELU -> hbuf -------
    F16x16 a1;
    #pragma unroll
    for (int j = 0; j < 16; ++j) {
      int K  = a_frag_k(lane, j);
      int Ks = (K < 6) ? K : 0;
      float v = inbuf[M * 6 + Ks];
      a1.h[j] = (K < 6) ? (_Float16)v : (_Float16)0.0f;
    }
    #pragma unroll
    for (int nt = 0; nt < 8; ++nt) {
      v16h bf = *(const v16h*)(smem + WB1_OFF + (size_t)(nt * 512 + lane * 16) * 2);
      v8f acc = {};
      acc = __builtin_amdgcn_wmma_f32_16x16x32_f16(false, a1.v, false, bf,
                                                   (short)0, acc, false, false);
      int Nc = nt * 16 + M;
      float bv = bias1[Nc];
      #pragma unroll
      for (int i = 0; i < 8; ++i)
        hbuf[(mrow + i) * 128 + Nc] = __float2half(gelu_exact(acc[i] + bv));
    }

    // ------- Layer 2: [16x128] @ [128x128] + b2, GELU -> hbuf -------
    F16x16 a2[4];
    #pragma unroll
    for (int kt = 0; kt < 4; ++kt) {
      #pragma unroll
      for (int p = 0; p < 8; ++p) {   // K and K+1 are adjacent halves in hbuf
        int K = a_frag_k(lane, 2 * p);
        a2[kt].u[p] = *(const unsigned*)&hbuf[M * 128 + kt * 32 + K];
      }
    }
    #pragma unroll
    for (int nt = 0; nt < 8; ++nt) {
      v8f acc = {};
      #pragma unroll
      for (int kt = 0; kt < 4; ++kt) {
        v16h bf = *(const v16h*)(smem + WB2_OFF +
                                 (size_t)((kt * 8 + nt) * 512 + lane * 16) * 2);
        acc = __builtin_amdgcn_wmma_f32_16x16x32_f16(false, a2[kt].v, false, bf,
                                                     (short)0, acc, false, false);
      }
      int Nc = nt * 16 + M;
      float bv = bias2[Nc];
      #pragma unroll
      for (int i = 0; i < 8; ++i)   // safe: all a2 fragments already in VGPRs
        hbuf[(mrow + i) * 128 + Nc] = __float2half(gelu_exact(acc[i] + bv));
    }

    // ------- Layer 3: [16x128] @ [128x64] + b3 -> kbuf (f32) -------
    F16x16 a3[4];
    #pragma unroll
    for (int kt = 0; kt < 4; ++kt) {
      #pragma unroll
      for (int p = 0; p < 8; ++p) {
        int K = a_frag_k(lane, 2 * p);
        a3[kt].u[p] = *(const unsigned*)&hbuf[M * 128 + kt * 32 + K];
      }
    }
    #pragma unroll
    for (int nt = 0; nt < 4; ++nt) {
      v8f acc = {};
      #pragma unroll
      for (int kt = 0; kt < 4; ++kt) {
        v16h bf = *(const v16h*)(smem + WB3_OFF +
                                 (size_t)((kt * 4 + nt) * 512 + lane * 16) * 2);
        acc = __builtin_amdgcn_wmma_f32_16x16x32_f16(false, a3[kt].v, false, bf,
                                                     (short)0, acc, false, false);
      }
      int Nc = nt * 16 + M;
      float bv = bias3[Nc];
      #pragma unroll
      for (int i = 0; i < 8; ++i)   // overwrites hbuf alias; a3 already in VGPRs
        kbuf[(mrow + i) * 64 + Nc] = acc[i] + bv;
    }

    // ------- out[node] = sum_m attn[m] * k[m,:] * f_y[nbr[m],:] -------
    const int c0 = lane << 1;       // 2 channels per lane (64 total)
    float o0 = 0.0f, o1 = 0.0f;
    #pragma unroll
    for (int m2 = 0; m2 < 16; ++m2) {
      float aw = attnb[m2];
      int   nb = nbrb[m2];
      float k0 = kbuf[m2 * 64 + c0];
      float k1 = kbuf[m2 * 64 + c0 + 1];
      const float2 fv = *(const float2*)&f_y[(size_t)nb * 64 + c0];
      o0 += aw * k0 * fv.x;
      o1 += aw * k1 * fv.y;
    }
    float2 res; res.x = o0; res.y = o1;
    *(float2*)&out[(size_t)node * 64 + c0] = res;
  }
}

extern "C" void kernel_launch(void* const* d_in, const int* in_sizes, int n_in,
                              void* d_out, int out_size, void* d_ws, size_t ws_size,
                              hipStream_t stream) {
  const float* y   = (const float*)d_in[0];
  const float* f_y = (const float*)d_in[1];
  const int*   nix = (const int*)d_in[2];
  // d_in[3] = neighbors_row_splits: fixed-degree CSR (deg = 16), not needed
  const float* W1 = (const float*)d_in[4];
  const float* b1 = (const float*)d_in[5];
  const float* W2 = (const float*)d_in[6];
  const float* b2 = (const float*)d_in[7];
  const float* W3 = (const float*)d_in[8];
  const float* b3 = (const float*)d_in[9];
  float* out = (float*)d_out;

  const int n = in_sizes[3] - 1;   // number of nodes

  // Allow > 64KB dynamic LDS (gfx1250 WGP has 320KB). Idempotent host call.
  (void)hipFuncSetAttribute((const void*)agno_fused_wmma,
                            hipFuncAttributeMaxDynamicSharedMemorySize,
                            (int)SMEM_BYTES);

  int blocks = (n + WAVES_PER_BLOCK - 1) / WAVES_PER_BLOCK;
  if (blocks > 1024) blocks = 1024;   // persistent blocks amortize weight staging
  agno_fused_wmma<<<dim3(blocks), dim3(THREADS), SMEM_BYTES, stream>>>(
      y, f_y, nix, W1, b1, W2, b2, W3, b3, out, n);
}
